// MSAM_87668872446688
// MI455X (gfx1250) — matrix operand order repaired and checked by reading, not compiled, not run on hardware
//
#include <hip/hip_runtime.h>
#include <hip/hip_bf16.h>

typedef __bf16 bf16;
typedef __attribute__((ext_vector_type(16))) __bf16 v16bf;
typedef __attribute__((ext_vector_type(8)))  __bf16 v8bf;
typedef __attribute__((ext_vector_type(8)))  float  v8f;
typedef __attribute__((ext_vector_type(4)))  float  v4f;

#define BATCH 8
#define C_OUT 256
#define C_HIGH 512
#define N_LOW 4096
#define N_HIGH 1024
#define PIX_HIGH 1024
#define DDIM 64
#define ESTRIDE 1028   /* f32 elems, padded row for energy tile  */
#define PSTRIDE 1032   /* bf16 elems, padded row for probs tile  */

// ---------------------------------------------------------------------------
// WMMA helpers: v_wmma_f32_16x16x32_bf16, D = A(16x32) * B(32x16) + C
// A lane layout (16-bit A 16x32): lane holds row M=lane&15; elements 0..7 =
//   K[off..off+7], elements 8..15 = K[16+off..16+off+7], off = (lane&16)?8:0.
// B lane layout (16-bit B 32x16): lane holds col N=lane&15; elements 0..7 =
//   K[off..off+7], elements 8..15 = K[16+off..16+off+7], off = (lane&16)?8:0.
// D layout: VGPR r -> row M = r + ((lane&16)?8:0), col N = lane&15.
// ---------------------------------------------------------------------------
__device__ __forceinline__ v8f wmma_bf16(v16bf a, v16bf b, v8f c) {
  return __builtin_amdgcn_wmma_f32_16x16x32_bf16(false, a, false, b,
                                                 (short)0, c, false, false);
}

__device__ __forceinline__ v16bf loadA_bf16(const bf16* __restrict__ row, int lane) {
  int off = (lane & 16) ? 8 : 0;
  v8bf lo = *(const v8bf*)(row + off);
  v8bf hi = *(const v8bf*)(row + off + 16);
  return __builtin_shufflevector(lo, hi, 0,1,2,3,4,5,6,7,8,9,10,11,12,13,14,15);
}

__device__ __forceinline__ v16bf loadB_bf16(const bf16* __restrict__ row, int lane) {
  int off = (lane & 16) ? 8 : 0;
  v8bf lo = *(const v8bf*)(row + off);
  v8bf hi = *(const v8bf*)(row + off + 16);
  return __builtin_shufflevector(lo, hi, 0,1,2,3,4,5,6,7,8,9,10,11,12,13,14,15);
}

// Streaming f32 -> bf16 A fragment: issue all 4 b128 loads, then convert.
__device__ __forceinline__ v16bf cvtA_f32(const float* __restrict__ row, int lane) {
  int off = (lane & 16) ? 8 : 0;
  v4f a = *(const v4f*)(row + off);
  v4f b = *(const v4f*)(row + off + 4);
  v4f c = *(const v4f*)(row + off + 16);
  v4f d = *(const v4f*)(row + off + 20);
  v16bf r;
#pragma unroll
  for (int i = 0; i < 4; ++i) {
    r[i]      = (bf16)a[i];
    r[i + 4]  = (bf16)b[i];
    r[i + 8]  = (bf16)c[i];
    r[i + 12] = (bf16)d[i];
  }
  return r;
}

// ---------------------------------------------------------------------------
// K_w0: w_conv f32 [256][512] -> bf16 with BN scale folded per output row.
// ---------------------------------------------------------------------------
__global__ __launch_bounds__(256) void k_cvt_wconv(
    const float* __restrict__ w, const float* __restrict__ gamma,
    const float* __restrict__ var, bf16* __restrict__ wbf) {
  int i = blockIdx.x * 256 + threadIdx.x;           // 131072 elements
  int o = i >> 9;
  float inv = gamma[o] * rsqrtf(var[o] + 1e-5f);
  wbf[i] = (bf16)(w[i] * inv);
}

// K_w1: generic f32 -> bf16 (fc1_w / fc2_w)
__global__ __launch_bounds__(256) void k_cvt_f32(
    const float* __restrict__ src, bf16* __restrict__ dst) {
  int i = blockIdx.x * 256 + threadIdx.x;
  dst[i] = (bf16)src[i];
}

// ---------------------------------------------------------------------------
// K0: fms_high [B][512][1024] f32  ->  fmsHT [B][1024][512] bf16 (transpose)
// ---------------------------------------------------------------------------
__global__ __launch_bounds__(256) void k_transpose_high(
    const float* __restrict__ in, bf16* __restrict__ out) {
  __shared__ bf16 tile[32][33];
  int b  = blockIdx.z;
  int p0 = blockIdx.x * 32;   // pixel tile
  int c0 = blockIdx.y * 32;   // channel tile
  const float* src = in + (size_t)b * C_HIGH * PIX_HIGH;
  bf16* dst = out + (size_t)b * PIX_HIGH * C_HIGH;
  int tx = threadIdx.x, ty = threadIdx.y;   // 32 x 8
#pragma unroll
  for (int i = 0; i < 4; ++i) {
    int c = c0 + ty + i * 8;
    tile[ty + i * 8][tx] = (bf16)src[(size_t)c * PIX_HIGH + p0 + tx];
  }
  __syncthreads();
#pragma unroll
  for (int i = 0; i < 4; ++i) {
    int p = p0 + ty + i * 8;
    dst[(size_t)p * C_HIGH + c0 + tx] = tile[tx][ty + i * 8];
  }
}

// ---------------------------------------------------------------------------
// K1: conv1x1 (BN scale pre-folded into weights) + bias + ReLU.
// xT[p,o] = sum_c fmsHT[p,c] * wbf[o,c]   (NT GEMM, all-bf16 operands)
// Writes featH twice: F row-major [1024][256] and FT col-major [256][1024],
// using the flat-view mapping F[n,c] = x[o=n>>2, p=(n&3)*256+c].
// One wave job = 16 p-rows x 64 o-cols; 2048 jobs = 256 blocks x 8 waves.
// ---------------------------------------------------------------------------
__global__ __launch_bounds__(256) void k_conv_bn_relu(
    const bf16* __restrict__ fmsHT, const bf16* __restrict__ wbf,
    const float* __restrict__ gamma, const float* __restrict__ beta,
    const float* __restrict__ mean, const float* __restrict__ var,
    bf16* __restrict__ F, bf16* __restrict__ FT) {
  int gtid = blockIdx.x * blockDim.x + threadIdx.x;
  int j    = gtid >> 5;          // wave job
  int lane = threadIdx.x & 31;
  int b    = j >> 8;
  int rem  = j & 255;
  int pt   = rem >> 2;           // p-tile 0..63
  int og   = rem & 3;            // o-group: o-tiles og*4 .. og*4+3

  const bf16* Ab = fmsHT + (size_t)b * PIX_HIGH * C_HIGH
                         + (size_t)(pt * 16 + (lane & 15)) * C_HIGH;
  const bf16* Bb0 = wbf + (size_t)((og * 4 + 0) * 16 + (lane & 15)) * C_HIGH;
  const bf16* Bb1 = wbf + (size_t)((og * 4 + 1) * 16 + (lane & 15)) * C_HIGH;
  const bf16* Bb2 = wbf + (size_t)((og * 4 + 2) * 16 + (lane & 15)) * C_HIGH;
  const bf16* Bb3 = wbf + (size_t)((og * 4 + 3) * 16 + (lane & 15)) * C_HIGH;
  v8f acc[4] = {};
#pragma unroll 4
  for (int ks = 0; ks < 16; ++ks) {           // K = 512
    v16bf a = loadA_bf16(Ab + ks * 32, lane);
    acc[0] = wmma_bf16(a, loadB_bf16(Bb0 + ks * 32, lane), acc[0]);
    acc[1] = wmma_bf16(a, loadB_bf16(Bb1 + ks * 32, lane), acc[1]);
    acc[2] = wmma_bf16(a, loadB_bf16(Bb2 + ks * 32, lane), acc[2]);
    acc[3] = wmma_bf16(a, loadB_bf16(Bb3 + ks * 32, lane), acc[3]);
  }

  bf16* Fb  = F  + (size_t)b * N_HIGH * C_OUT;
  bf16* FTb = FT + (size_t)b * C_OUT * N_HIGH;
  int hi = (lane & 16) ? 8 : 0;
#pragma unroll
  for (int t = 0; t < 4; ++t) {
    int o = (og * 4 + t) * 16 + (lane & 15);
    float inv = gamma[o] * rsqrtf(var[o] + 1e-5f);
    float bia = beta[o] - mean[o] * inv;      // scale already folded in wbf
    int pbase = pt * 16 + hi;                 // 8 consecutive pixels
    int n = 4 * o + (pbase >> 8);             // constant across r
    int cb = pbase & 255;                     // c = cb + r
    v8bf fv;
#pragma unroll
    for (int r = 0; r < 8; ++r) {
      float v = fmaxf(acc[t][r] + bia, 0.0f);
      fv[r] = (bf16)v;
      FTb[(size_t)(cb + r) * N_HIGH + n] = fv[r];
    }
    *(v8bf*)(Fb + (size_t)n * C_OUT + cb) = fv;
  }
}

// ---------------------------------------------------------------------------
// K2: mid_low[n,d] = featL[n,:] . fc1_w[d,:] + fc1_b[d]   (featL = flat fms_low)
// One wave job = 16 n-rows x all 64 d; 2048 jobs.
// ---------------------------------------------------------------------------
__global__ __launch_bounds__(256) void k_mid_low(
    const float* __restrict__ fms_low, const bf16* __restrict__ fc1bf,
    const float* __restrict__ fc1_b, bf16* __restrict__ midL) {
  int gtid = blockIdx.x * blockDim.x + threadIdx.x;
  int j    = gtid >> 5;
  int lane = threadIdx.x & 31;
  int b  = j >> 8;
  int nt = j & 255;

  const float* Ab = fms_low + (size_t)b * N_LOW * C_OUT
                            + (size_t)(nt * 16 + (lane & 15)) * C_OUT;
  const bf16* Bb0 = fc1bf + (size_t)( 0 + (lane & 15)) * C_OUT;
  const bf16* Bb1 = fc1bf + (size_t)(16 + (lane & 15)) * C_OUT;
  const bf16* Bb2 = fc1bf + (size_t)(32 + (lane & 15)) * C_OUT;
  const bf16* Bb3 = fc1bf + (size_t)(48 + (lane & 15)) * C_OUT;
  v8f acc[4] = {};
#pragma unroll 2
  for (int ks = 0; ks < 8; ++ks) {            // K = 256
    v16bf a = cvtA_f32(Ab + ks * 32, lane);
    acc[0] = wmma_bf16(a, loadB_bf16(Bb0 + ks * 32, lane), acc[0]);
    acc[1] = wmma_bf16(a, loadB_bf16(Bb1 + ks * 32, lane), acc[1]);
    acc[2] = wmma_bf16(a, loadB_bf16(Bb2 + ks * 32, lane), acc[2]);
    acc[3] = wmma_bf16(a, loadB_bf16(Bb3 + ks * 32, lane), acc[3]);
  }
  bf16* Ob = midL + (size_t)b * N_LOW * DDIM;
  int hi = (lane & 16) ? 8 : 0;
#pragma unroll
  for (int t = 0; t < 4; ++t) {
    int d = t * 16 + (lane & 15);
    float bia = fc1_b[d];
#pragma unroll
    for (int r = 0; r < 8; ++r) {
      int n = nt * 16 + hi + r;
      Ob[(size_t)n * DDIM + d] = (bf16)(acc[t][r] + bia);
    }
  }
}

// ---------------------------------------------------------------------------
// K3: mid_high[m,d] = F[m,:] . fc2_w[d,:] + fc2_b[d].  512 wave jobs.
// ---------------------------------------------------------------------------
__global__ __launch_bounds__(256) void k_mid_high(
    const bf16* __restrict__ F, const bf16* __restrict__ fc2bf,
    const float* __restrict__ fc2_b, bf16* __restrict__ midH) {
  int gtid = blockIdx.x * blockDim.x + threadIdx.x;
  int j    = gtid >> 5;
  int lane = threadIdx.x & 31;
  int b  = j >> 6;
  int nt = j & 63;

  const bf16* Ab = F + (size_t)b * N_HIGH * C_OUT
                     + (size_t)(nt * 16 + (lane & 15)) * C_OUT;
  const bf16* Bb0 = fc2bf + (size_t)( 0 + (lane & 15)) * C_OUT;
  const bf16* Bb1 = fc2bf + (size_t)(16 + (lane & 15)) * C_OUT;
  const bf16* Bb2 = fc2bf + (size_t)(32 + (lane & 15)) * C_OUT;
  const bf16* Bb3 = fc2bf + (size_t)(48 + (lane & 15)) * C_OUT;
  v8f acc[4] = {};
#pragma unroll 2
  for (int ks = 0; ks < 8; ++ks) {            // K = 256
    v16bf a = loadA_bf16(Ab + ks * 32, lane);
    acc[0] = wmma_bf16(a, loadB_bf16(Bb0 + ks * 32, lane), acc[0]);
    acc[1] = wmma_bf16(a, loadB_bf16(Bb1 + ks * 32, lane), acc[1]);
    acc[2] = wmma_bf16(a, loadB_bf16(Bb2 + ks * 32, lane), acc[2]);
    acc[3] = wmma_bf16(a, loadB_bf16(Bb3 + ks * 32, lane), acc[3]);
  }
  bf16* Ob = midH + (size_t)b * N_HIGH * DDIM;
  int hi = (lane & 16) ? 8 : 0;
#pragma unroll
  for (int t = 0; t < 4; ++t) {
    int d = t * 16 + (lane & 15);
    float bia = fc2_b[d];
#pragma unroll
    for (int r = 0; r < 8; ++r) {
      int n = nt * 16 + hi + r;
      Ob[(size_t)n * DDIM + d] = (bf16)(acc[t][r] + bia);
    }
  }
}

// ---------------------------------------------------------------------------
// K4: fused attention.  Per workgroup: 32 attention rows (n) of one batch.
//   Phase 1: energy[32,1024] = midL_tile @ midH^T  -> LDS (f32, padded rows)
//   Phase 2: softmax over m=1024 (shfl reductions) -> probs bf16 in LDS
//   Phase 3: out^T[n,c] = probs @ F  via B = FT[c,:] (contiguous), then store
//            transposed: out[c*4096 + n] = scale * val  (two float4 per lane)
// ---------------------------------------------------------------------------
__global__ __launch_bounds__(256) void k_attn(
    const bf16* __restrict__ midL, const bf16* __restrict__ midH,
    const bf16* __restrict__ FT, const float* __restrict__ scale,
    float* __restrict__ out) {
  extern __shared__ char smem[];
  float* Es = (float*)smem;                                  // [32][ESTRIDE]
  bf16*  Ps = (bf16*)(smem + 32 * ESTRIDE * sizeof(float));  // [32][PSTRIDE]

  int b  = blockIdx.x >> 7;
  int nt = blockIdx.x & 127;
  int n0 = nt * 32;
  int tid  = threadIdx.x;
  int w    = tid >> 5;
  int lane = tid & 31;
  int hi   = (lane & 16) ? 8 : 0;

  // ---- Phase 1: energy ----
  {
    int ns = w & 1;                       // n sub-tile (rows ns*16..ns*16+15)
    const bf16* Lrow = midL + ((size_t)b * N_LOW + n0 + ns * 16 + (lane & 15)) * DDIM;
    v16bf a0 = loadA_bf16(Lrow, lane);
    v16bf a1 = loadA_bf16(Lrow + 32, lane);
    const bf16* Hb = midH + (size_t)b * N_HIGH * DDIM;
#pragma unroll 4
    for (int jj = 0; jj < 16; ++jj) {
      int mt = (w >> 1) + jj * 4;         // m-tile 0..63
      const bf16* Hrow = Hb + (size_t)(mt * 16 + (lane & 15)) * DDIM;
      v8f e = {};
      e = wmma_bf16(a0, loadB_bf16(Hrow, lane), e);
      e = wmma_bf16(a1, loadB_bf16(Hrow + 32, lane), e);
      int m  = mt * 16 + (lane & 15);
      int nl = ns * 16 + hi;
#pragma unroll
      for (int r = 0; r < 8; ++r) Es[(size_t)(nl + r) * ESTRIDE + m] = e[r];
    }
  }
  __syncthreads();

  // ---- Phase 2: softmax over m (each wave: 4 rows) ----
#pragma unroll
  for (int rr = 0; rr < 4; ++rr) {
    int row = w * 4 + rr;
    float* Er = Es + (size_t)row * ESTRIDE;
    float mx = -3.0e38f;
    for (int i = lane; i < N_HIGH; i += 32) mx = fmaxf(mx, Er[i]);
#pragma unroll
    for (int o = 16; o > 0; o >>= 1) mx = fmaxf(mx, __shfl_xor(mx, o, 32));
    float s = 0.0f;
    for (int i = lane; i < N_HIGH; i += 32) {
      float ex = __expf(Er[i] - mx);
      Er[i] = ex;
      s += ex;
    }
#pragma unroll
    for (int o = 16; o > 0; o >>= 1) s += __shfl_xor(s, o, 32);
    float inv = 1.0f / s;
    bf16* Pr = Ps + (size_t)row * PSTRIDE;
    for (int i = lane; i < N_HIGH; i += 32) Pr[i] = (bf16)(Er[i] * inv);
  }
  __syncthreads();

  // ---- Phase 3: out = probs @ F  (B operand = FT rows, contiguous) ----
  float sc = scale[0];
  const bf16* FTb = FT + (size_t)b * C_OUT * N_HIGH;
  float* outb = out + (size_t)b * C_OUT * N_LOW;
#pragma unroll
  for (int q = 0; q < 2; ++q) {
    int ct = w * 2 + q;                   // c-tile 0..15
    const bf16* Brow = FTb + (size_t)(ct * 16 + (lane & 15)) * N_HIGH;
    const bf16* P0   = Ps + (size_t)(lane & 15) * PSTRIDE;
    v8f acc0 = {}, acc1 = {};
#pragma unroll 4
    for (int ks = 0; ks < 32; ++ks) {     // K = 1024
      v16bf bb  = loadB_bf16(Brow + ks * 32, lane);
      v16bf pa0 = loadA_bf16(P0 + ks * 32, lane);
      v16bf pa1 = loadA_bf16(P0 + (size_t)16 * PSTRIDE + ks * 32, lane);
      acc0 = wmma_bf16(pa0, bb, acc0);    // rows n0+0..15
      acc1 = wmma_bf16(pa1, bb, acc1);    // rows n0+16..31
    }
    int c = ct * 16 + (lane & 15);
    float* o0 = outb + (size_t)c * N_LOW + n0 + hi;   // ns=0, 8 consecutive n
    float* o1 = o0 + 16;                              // ns=1
    v4f x;
    x[0]=acc0[0]*sc; x[1]=acc0[1]*sc; x[2]=acc0[2]*sc; x[3]=acc0[3]*sc;
    *(v4f*)o0 = x;
    x[0]=acc0[4]*sc; x[1]=acc0[5]*sc; x[2]=acc0[6]*sc; x[3]=acc0[7]*sc;
    *(v4f*)(o0 + 4) = x;
    x[0]=acc1[0]*sc; x[1]=acc1[1]*sc; x[2]=acc1[2]*sc; x[3]=acc1[3]*sc;
    *(v4f*)o1 = x;
    x[0]=acc1[4]*sc; x[1]=acc1[5]*sc; x[2]=acc1[6]*sc; x[3]=acc1[7]*sc;
    *(v4f*)(o1 + 4) = x;
  }
}

// ---------------------------------------------------------------------------
extern "C" void kernel_launch(void* const* d_in, const int* in_sizes, int n_in,
                              void* d_out, int out_size, void* d_ws, size_t ws_size,
                              hipStream_t stream) {
  const float* fms_low  = (const float*)d_in[0];
  const float* fms_high = (const float*)d_in[1];
  const float* w_conv   = (const float*)d_in[2];
  const float* bn_gamma = (const float*)d_in[3];
  const float* bn_beta  = (const float*)d_in[4];
  const float* bn_mean  = (const float*)d_in[5];
  const float* bn_var   = (const float*)d_in[6];
  const float* fc1_w    = (const float*)d_in[7];
  const float* fc1_b    = (const float*)d_in[8];
  const float* fc2_w    = (const float*)d_in[9];
  const float* fc2_b    = (const float*)d_in[10];
  const float* scale    = (const float*)d_in[11];
  float* out = (float*)d_out;

  char* ws = (char*)d_ws;
  bf16* fmsHT = (bf16*)(ws);                    //  8,388,608 B
  bf16* F     = (bf16*)(ws + 8388608);          //  4,194,304 B
  bf16* FT    = (bf16*)(ws + 12582912);         //  4,194,304 B
  bf16* midL  = (bf16*)(ws + 16777216);         //  4,194,304 B
  bf16* midH  = (bf16*)(ws + 20971520);         //  1,048,576 B
  bf16* wcbf  = (bf16*)(ws + 22020096);         //    262,144 B
  bf16* fc1bf = (bf16*)(ws + 22282240);         //     32,768 B
  bf16* fc2bf = (bf16*)(ws + 22315008);         //     32,768 B  (~22.3 MB)

  // Weight conversions (once, tiny)
  k_cvt_wconv<<<512, 256, 0, stream>>>(w_conv, bn_gamma, bn_var, wcbf);
  k_cvt_f32<<<64, 256, 0, stream>>>(fc1_w, fc1bf);
  k_cvt_f32<<<64, 256, 0, stream>>>(fc2_w, fc2bf);

  k_transpose_high<<<dim3(32, 16, BATCH), dim3(32, 8), 0, stream>>>(fms_high, fmsHT);
  k_conv_bn_relu<<<256, 256, 0, stream>>>(fmsHT, wcbf, bn_gamma, bn_beta,
                                          bn_mean, bn_var, F, FT);
  k_mid_low<<<256, 256, 0, stream>>>(fms_low, fc1bf, fc1_b, midL);
  k_mid_high<<<64, 256, 0, stream>>>(F, fc2bf, fc2_b, midH);

  size_t smem = 32 * ESTRIDE * sizeof(float) + 32 * PSTRIDE * sizeof(bf16);
  k_attn<<<BATCH * 128, 256, smem, stream>>>(midL, midH, FT, scale, out);
}
